// TNTCN_8151847928343
// MI455X (gfx1250) — matrix operand order repaired and checked by reading, compiled, not run on hardware
//
#include <hip/hip_runtime.h>
#include <hip/hip_bf16.h>

// ---------------------------------------------------------------------------
// MI455X (gfx1250) TCN implementation.
// fp32 storage in HBM (exact residual chain), f16 V_WMMA_F32_16X16X32_F16 for
// all conv GEMMs. Weights pre-packed on device into per-lane WMMA A-fragment
// layout (wave32): A frag half h of lane l (m=l&15, hi=l>>4) holds
//   K = hi*16+h (same contiguous order as B)
// B frag half h of lane l (n=l&15) holds K = hi*16+h  -> one 32B LDS load.
// C/D f32: VGPR v of lane l holds row m = v+8*(l>>4), col n = l&15.
// Input windows are staged with CDNA5 async copies
// (global_load_async_to_lds_b128, ASYNCcnt) -- no VGPR round-trip.
// ---------------------------------------------------------------------------

typedef __attribute__((ext_vector_type(16))) _Float16 v16h;
typedef __attribute__((ext_vector_type(8)))  float    v8f;

#define BB   16
#define CINC 16
#define NNODE 2048
#define TFULL 96
#define TCCH 32
#define T1   84
#define T2   72
#define OUTS 12
#define KINC 224     /* unified inception K = 32 ch * 7 tap slots */
#define KEND 1952    /* 32 ch * 61 taps */

struct P { const float* p[40]; };
// d_in flatten order (setup_inputs dict order):
// 0:x 1:channel_w 2:channel_b
// 3..10  : filter_w[l][k]  (l=0..1, k=0..3)
// 11..18 : filter_b[l][k]
// 19..26 : gate_w[l][k]
// 27..34 : gate_b[l][k]
// 35:end_w 36:end_b 37:out_w 38:out_b

// Async 16B global -> LDS copy (CDNA5 TDM-lite path, ASYNCcnt tracked).
__device__ __forceinline__ void async_ld16(unsigned ldsOff, const void* saddr,
                                           unsigned voff) {
  asm volatile("global_load_async_to_lds_b128 %0, %1, %2 offset:0"
               :: "v"(ldsOff), "v"(voff), "s"(saddr) : "memory");
}
__device__ __forceinline__ void async_wait0() {
  asm volatile("s_wait_asynccnt 0x0" ::: "memory");
}

// -------------------------- weight packing --------------------------------

__global__ void k_pack_a1(P pp, _Float16* a1) {
  int idx = blockIdx.x * 256 + threadIdx.x;            // [tile2][lane32][h16]
  if (idx >= 2 * 32 * 16) return;
  int tile = idx >> 9, lane = (idx >> 4) & 31, h = idx & 15;
  int m = lane & 15, hi = lane >> 4;
  int K = hi * 16 + h;
  int oc = tile * 16 + m;
  float v = (K < CINC) ? pp.p[1][oc * CINC + K] : 0.0f;  // channel_w (32,16,1,1)
  a1[idx] = (_Float16)v;
}

__global__ void k_pack_a2(P pp, _Float16* a2) {
  int idx = blockIdx.x * 256 + threadIdx.x;  // [layer2][tile4][s7][lane32][h16]
  if (idx >= 2 * 4 * 7 * 32 * 16) return;
  int layer = idx / 14336;
  int r = idx % 14336;
  int tile = r / 3584;           // 0:f rows0-15  1:f rows16-31  2,3: gate
  int r2 = r % 3584;
  int s = r2 >> 9;
  int lane = (r2 >> 4) & 31, h = r2 & 15;
  int m = lane & 15, hi = lane >> 4;
  int kk = s * 32 + hi * 16 + h;
  int fg = tile >> 1;
  int oc = (tile & 1) * 16 + m;
  int branch = oc >> 3, mm = oc & 7;
  const int KSv[4] = {2, 3, 6, 7};
  int kb = KSv[branch];
  int c = kk / 7, d = kk % 7;    // tap slot d -> input offset 2d (right-aligned)
  int j = d - (7 - kb);          // tap index inside the branch kernel
  float v = 0.0f;
  if (j >= 0 && j < kb) {
    const float* w = pp.p[(fg ? 19 : 3) + layer * 4 + branch]; // (8,32,1,kb)
    v = w[(mm * 32 + c) * kb + j];
  }
  a2[idx] = (_Float16)v;
}

__global__ void k_pack_ae(P pp, _Float16* ae) {
  int idx = blockIdx.x * 256 + threadIdx.x;  // [tile2][s61][lane32][h16]
  if (idx >= 2 * 61 * 32 * 16) return;
  int tile = idx / 31232;
  int r = idx % 31232;
  int s = r >> 9;
  int lane = (r >> 4) & 31, h = r & 15;
  int m = lane & 15, hi = lane >> 4;
  int kk = s * 32 + hi * 16 + h;
  int c = kk / 61, j = kk % 61;
  float v = pp.p[35][((tile * 16 + m) * 32 + c) * 61 + j];  // end_w (32,32,1,61)
  ae[idx] = (_Float16)v;
}

__global__ void k_pack_bias(P pp, float* bias2) {
  int idx = threadIdx.x;                 // [layer2][fg2*32]
  if (idx >= 128) return;
  int layer = idx >> 6, r = idx & 63;
  int fg = r >> 5, oc = r & 31;
  int branch = oc >> 3, mm = oc & 7;
  bias2[idx] = pp.p[(fg ? 27 : 11) + layer * 4 + branch][mm];
}

// -------------------------- stage 1: 1x1 channel conv ----------------------
// GEMM [32oc x 16cin] x [16 x t]; block = 4 waves covering 32 t of one (b,n).

__global__ void k_stage1(const float* __restrict__ x, const float* __restrict__ cb,
                         const _Float16* __restrict__ a1, float* __restrict__ hbuf) {
  __shared__ __attribute__((aligned(32))) _Float16 sBX[32 * 32];  // [tcol][K32]
  const int bn = blockIdx.y;
  const int b = bn >> 11, n = bn & (NNODE - 1);
  const int t0 = blockIdx.x * 32;
  const int tid = threadIdx.x;

  for (int idx = tid; idx < 32 * 32; idx += 128) {
    int tcol = idx >> 5, c = idx & 31;
    float v = 0.0f;
    if (c < CINC) v = x[((b * CINC + c) * NNODE + n) * TFULL + t0 + tcol];
    sBX[tcol * 32 + c] = (_Float16)v;
  }
  __syncthreads();

  const int wave = tid >> 5, lane = tid & 31;
  const int tile = wave & 1, tsub = wave >> 1;
  const int ncol = lane & 15, hi = lane >> 4;

  v16h a = *(const v16h*)(a1 + (tile * 32 + lane) * 16);
  v16h bf = *(const v16h*)(sBX + (tsub * 16 + ncol) * 32 + hi * 16);
  v8f acc = {};
  acc = __builtin_amdgcn_wmma_f32_16x16x32_f16(false, a, false, bf,
                                               (short)0, acc, false, false);
#pragma unroll
  for (int vv = 0; vv < 8; ++vv) {
    int oc = tile * 16 + vv + 8 * hi;
    hbuf[(bn * TCCH + oc) * TFULL + t0 + tsub * 16 + ncol] = acc[vv] + cb[oc];
  }
}

// ----------------- stage 2: gated dilated inception + residual -------------
// Unified K=224 GEMM for filter AND gate. 256 threads = 8 waves:
// 4 M-tiles (f-lo,f-hi,g-lo,g-hi) x 2 N-subtiles = 32 output columns/block.
// Window (32ch x 44t fp32) staged via async global->LDS copies.

__global__ void k_stage2(const float* __restrict__ in, int Tin,
                         const float* __restrict__ origin, int resOff,
                         const _Float16* __restrict__ apack,   // [4][7][32][16]
                         const float* __restrict__ bias64,
                         float* __restrict__ out, int Tout) {
  __shared__ __attribute__((aligned(16))) float sWinF[32 * 44];  // 5632 B
  __shared__ __attribute__((aligned(32))) _Float16 sB[32 * KINC];// 14336 B
  __shared__ float sD[64 * 32];                                  // 8192 B
  const int bn = blockIdx.y;
  const int t0 = blockIdx.x * 32;
  const int tid = threadIdx.x;   // 256

  // --- async stage: 32 rows x 44 floats (11 x 16B chunks per row) ---
  // Tail tiles read past the row end; that data only reaches output columns
  // >= Tout (never stored) and stays inside the contiguous workspace.
  {
    const char* gbase = (const char*)(in + (size_t)bn * TCCH * Tin + t0);
    unsigned ldsBase = (unsigned)(size_t)&sWinF[0];
    for (int chunk = tid; chunk < 32 * 11; chunk += 256) {
      int c = chunk / 11, q = chunk % 11;
      async_ld16(ldsBase + (unsigned)(c * 44 * 4 + q * 16), gbase,
                 (unsigned)(c * Tin * 4 + q * 16));
    }
    async_wait0();
  }
  __syncthreads();

  // --- build column-major B (f16): sB[ncol][kk], kk=(c,d) -> in[c][ncol+2d]
  for (int idx = tid; idx < 32 * KINC; idx += 256) {
    int ncol = idx / KINC, kk = idx % KINC;
    int c = kk / 7, d = kk % 7;
    sB[idx] = (_Float16)sWinF[c * 44 + ncol + 2 * d];
  }
  __syncthreads();

  const int wave = tid >> 5, lane = tid & 31;
  const int mtile = wave & 3, nsub = wave >> 2;
  const int n16 = lane & 15, hi = lane >> 4;
  const _Float16* ap = apack + (mtile * 7 * 32 + lane) * 16;
  const _Float16* bp = sB + (nsub * 16 + n16) * KINC + hi * 16;
  v8f acc = {};
#pragma unroll
  for (int s = 0; s < 7; ++s) {
    v16h a = *(const v16h*)(ap + s * 32 * 16);
    v16h b = *(const v16h*)(bp + s * 32);
    acc = __builtin_amdgcn_wmma_f32_16x16x32_f16(false, a, false, b,
                                                 (short)0, acc, false, false);
  }
  {
    int fg = mtile >> 1;
    int ocb = (mtile & 1) * 16;
#pragma unroll
    for (int vv = 0; vv < 8; ++vv) {
      int oc = ocb + vv + 8 * hi;
      sD[(fg * 32 + oc) * 32 + nsub * 16 + n16] = acc[vv] + bias64[fg * 32 + oc];
    }
  }
  __syncthreads();

  // --- fused tanh * sigmoid + residual-to-origin, masked store ---
  for (int idx = tid; idx < 32 * 32; idx += 256) {
    int oc = idx >> 5, nc = idx & 31;
    int t = t0 + nc;
    if (t < Tout) {
      float f = sD[oc * 32 + nc];
      float g = sD[(32 + oc) * 32 + nc];
      float val = tanhf(f) * (1.0f / (1.0f + expf(-g)))
                + origin[(bn * TCCH + oc) * TFULL + t + resOff];
      out[(bn * TCCH + oc) * Tout + t] = val;
    }
  }
}

// -------------- stage 3: end conv (K=1952) + ELU + 1x1 out conv ------------

__global__ void k_stage3(const float* __restrict__ cur2, const _Float16* __restrict__ ae,
                         const float* __restrict__ end_b, const float* __restrict__ out_w,
                         const float* __restrict__ out_b, float* __restrict__ y) {
  __shared__ __attribute__((aligned(16))) float sWinF[32 * T2];    // 9216 B
  __shared__ __attribute__((aligned(32))) _Float16 sB[12 * KEND];  // 46848 B
  __shared__ float sE[32 * 16];
  const int bn = blockIdx.x;
  const int tid = threadIdx.x;   // 64 threads = 2 waves

  // --- async stage: whole (32 x 72) fp32 slice = 9216 B = 576 x 16B ---
  {
    const char* gbase = (const char*)(cur2 + (size_t)bn * TCCH * T2);
    unsigned ldsBase = (unsigned)(size_t)&sWinF[0];
    for (int chunk = tid; chunk < 576; chunk += 64) {
      unsigned off = (unsigned)(chunk * 16);
      async_ld16(ldsBase + off, gbase, off);
    }
    async_wait0();
  }
  __syncthreads();

  for (int idx = tid; idx < 12 * KEND; idx += 64) {
    int ncol = idx / KEND, kk = idx % KEND;
    int c = kk / 61, j = kk % 61;
    sB[idx] = (_Float16)sWinF[c * T2 + ncol + j];
  }
  __syncthreads();

  const int wave = tid >> 5, lane = tid & 31;
  const int ncol = lane & 15, hi = lane >> 4;
  const int bcol = (ncol < OUTS) ? ncol : 0;     // cols 12..15 compute garbage, never stored
  const _Float16* ap = ae + (wave * 61 * 32 + lane) * 16;
  const _Float16* bp = sB + bcol * KEND + hi * 16;
  v8f acc = {};
  for (int s = 0; s < 61; ++s) {
    v16h a = *(const v16h*)(ap + s * 32 * 16);
    v16h b = *(const v16h*)(bp + s * 32);
    acc = __builtin_amdgcn_wmma_f32_16x16x32_f16(false, a, false, b,
                                                 (short)0, acc, false, false);
  }
#pragma unroll
  for (int vv = 0; vv < 8; ++vv) {
    int oc = wave * 16 + vv + 8 * hi;
    float e = acc[vv] + end_b[oc];
    e = (e > 0.0f) ? e : expm1f(e);              // ELU
    sE[oc * 16 + ncol] = e;
  }
  __syncthreads();
  if (tid < OUTS) {
    float a = out_b[0];
#pragma unroll
    for (int oc = 0; oc < 32; ++oc) a += out_w[oc] * sE[oc * 16 + tid];
    int b = bn >> 11, n = bn & (NNODE - 1);
    y[(b * OUTS + tid) * NNODE + n] = a;         // (B, OUT_SEQ, N)
  }
}

// ------------------------------- launch ------------------------------------

extern "C" void kernel_launch(void* const* d_in, const int* in_sizes, int n_in,
                              void* d_out, int out_size, void* d_ws, size_t ws_size,
                              hipStream_t stream) {
  (void)in_sizes; (void)out_size; (void)ws_size;
  const int BN = BB * NNODE;

  char* ws = (char*)d_ws;
  const size_t szH  = (size_t)BN * TCCH * TFULL * 4;  // 402,653,184
  const size_t szC1 = (size_t)BN * TCCH * T1 * 4;     // 352,321,536
  const size_t szC2 = (size_t)BN * TCCH * T2 * 4;     // 301,989,888
  float* hbuf = (float*)ws;
  float* cur1 = (float*)(ws + szH);
  float* cur2 = (float*)(ws + szH + szC1);
  char*  wb   = ws + szH + szC1 + szC2;
  _Float16* a1 = (_Float16*)wb;                        // 2 KB
  _Float16* a2 = (_Float16*)(wb + 4096);               // 56 KB
  _Float16* ae = (_Float16*)(wb + 4096 + 57344);       // 122 KB
  float* bias2 = (float*)(wb + 4096 + 57344 + 124928); // 512 B

  P pp;
  for (int i = 0; i < 40; ++i) pp.p[i] = (i < n_in) ? (const float*)d_in[i] : nullptr;

  k_pack_a1<<<(1024 + 255) / 256, 256, 0, stream>>>(pp, a1);
  k_pack_a2<<<(28672 + 255) / 256, 256, 0, stream>>>(pp, a2);
  k_pack_ae<<<(62464 + 255) / 256, 256, 0, stream>>>(pp, ae);
  k_pack_bias<<<1, 128, 0, stream>>>(pp, bias2);

  k_stage1<<<dim3(3, BN), 128, 0, stream>>>((const float*)d_in[0],
                                            (const float*)d_in[2], a1, hbuf);
  // layer 0: 96 -> 84, residual offset 12 (32-col tiles)
  k_stage2<<<dim3(3, BN), 256, 0, stream>>>(hbuf, TFULL, hbuf, 12,
                                            a2, bias2, cur1, T1);
  // layer 1: 84 -> 72, residual offset 24
  k_stage2<<<dim3(3, BN), 256, 0, stream>>>(cur1, T1, hbuf, 24,
                                            a2 + 14336, bias2 + 64, cur2, T2);
  k_stage3<<<BN, 64, 0, stream>>>(cur2, ae, (const float*)d_in[36],
                                  (const float*)d_in[37], (const float*)d_in[38],
                                  (float*)d_out);
}
